// TransformerBlock_45397804318794
// MI455X (gfx1250) — matrix operand order repaired and checked by reading, not compiled, not run on hardware
//
#include <hip/hip_runtime.h>
#include <math.h>

typedef __bf16 bf16;
typedef unsigned int u32;
typedef __attribute__((ext_vector_type(16))) __bf16 v16bf;
typedef __attribute__((ext_vector_type(8)))  float  v8f;
typedef __attribute__((ext_vector_type(4)))  u32    v4u;

#define B_   2
#define E_   3
#define F_   128
#define H_   128
#define W_   128
#define HW_  (H_*W_)
#define NH_  8
#define WSZ  8
#define SS_  4
#define NW_  256      /* (H/WS)*(W/WS) */
#define NT_  64       /* tokens per window */
#define DH_  16       /* head dim = F/NH */
#define TOK_ 98304    /* B*E*NW*NT */
#define PIX_ 32768    /* B*H*W */
#define C3_  384
#define C6_  768
#define C12_ 1536

#define AKP  40       /* padded row length (elems) of As tile   */
#define BKP  40       /* padded row length (elems) of Bs[n][k]  */
#define PSP  72       /* padded row length of Ps (softmax probs) */
#define VTP  72       /* padded row length of Vt (transposed V)  */

__constant__ int d_QE[6] = {1,1,0,0,2,2};
__constant__ int d_KE[6] = {0,2,1,2,1,0};

__device__ inline float gelu_exact(float x) {
  return 0.5f * x * (1.0f + erff(x * 0.70710678118654752f));
}

union frag_cvt { v4u u[2]; v16bf v; };

// ---------------------------------------------------------------------------
// Generic tiled GEMM: C[MxN] = A[MxK](bf16,row-major) * B[KxN](bf16,row-major)
// (+bias). Output f32 (Cf!=null) or bf16 (Cb). K multiple of 32, M,N even.
// Block = 256 threads = 8 wave32 waves; tile 64(M) x 128(N).
// A staged row-major (padded); B staged TRANSPOSED (Bs[n][k], padded) so each
// WMMA fragment is two ds_load_b128. Fragment layouts per CDNA5 ISA 7.12.2.
// ---------------------------------------------------------------------------
__global__ __launch_bounds__(256) void gemm_bf16_kernel(
    const bf16* __restrict__ A, const bf16* __restrict__ Bm,
    const float* __restrict__ bias, float* __restrict__ Cf,
    bf16* __restrict__ Cb, int M, int N, int K)
{
  __shared__ bf16 As[64 * AKP];
  __shared__ bf16 Bs[128 * BKP];
  const int m0   = blockIdx.y * 64;
  const int n0   = blockIdx.x * 128;
  const int tid  = threadIdx.x;
  const int lane = tid & 31;
  const int wave = tid >> 5;   // 0..7
  const int wm   = wave >> 1;  // 0..3 : 16-row slice
  const int wn   = wave & 1;   // 0..1 : 64-col slice

  v8f acc[4] = {};

  for (int k0 = 0; k0 < K; k0 += 32) {
    // --- stage A tile (64x32): one b128 load + one b128 LDS store per thread
    {
      int r  = tid >> 2;          // 0..63
      int cc = (tid & 3) * 8;     // 0,8,16,24
      int gm = m0 + r;
      v4u v = {};
      if (gm < M) v = *reinterpret_cast<const v4u*>(A + (size_t)gm * K + k0 + cc);
      *reinterpret_cast<v4u*>(As + r * AKP + cc) = v;
    }
    // --- stage B tile (32x128) transposed into Bs[n][k] via 2x2 reg transpose
    {
      int np = tid & 63;          // n-pair: n = 2*np
      int kb = tid >> 6;          // 0..3
      int n  = 2 * np;
      int gn = n0 + n;
      #pragma unroll
      for (int i = 0; i < 4; ++i) {
        int k = 2 * (kb + 4 * i); // k-pair: k, k+1 (0..30)
        u32 r0 = 0u, r1 = 0u;
        if (gn < N) {
          r0 = *reinterpret_cast<const u32*>(Bm + (size_t)(k0 + k) * N + gn);
          r1 = *reinterpret_cast<const u32*>(Bm + (size_t)(k0 + k + 1) * N + gn);
        }
        u32 c0 = (r0 & 0xffffu) | (r1 << 16);
        u32 c1 = (r0 >> 16) | (r1 & 0xffff0000u);
        *reinterpret_cast<u32*>(Bs + n * BKP + k)       = c0;
        *reinterpret_cast<u32*>(Bs + (n + 1) * BKP + k) = c1;
      }
    }
    if (k0 + 32 < K) {
      __builtin_prefetch(A + (size_t)(m0 + (tid & 63)) * K + k0 + 32, 0, 1);
      __builtin_prefetch(Bm + (size_t)(k0 + 32 + (tid & 31)) * N + n0, 0, 1);
    }
    __syncthreads();

    // A fragment: two b128 LDS loads
    v16bf af;
    {
      int m  = wm * 16 + (lane & 15);
      int kh = lane >> 4;
      const bf16* p = As + m * AKP + kh * 8;
      frag_cvt c;
      c.u[0] = *reinterpret_cast<const v4u*>(p);
      c.u[1] = *reinterpret_cast<const v4u*>(p + 16);
      af = c.v;
    }
    #pragma unroll
    for (int nt = 0; nt < 4; ++nt) {
      int n = wn * 64 + nt * 16 + (lane & 15);
      const bf16* p = Bs + n * BKP + (lane >> 4) * 8;
      frag_cvt c;
      c.u[0] = *reinterpret_cast<const v4u*>(p);
      c.u[1] = *reinterpret_cast<const v4u*>(p + 16);
      acc[nt] = __builtin_amdgcn_wmma_f32_16x16x32_bf16(
          false, af, false, c.v, (short)0, acc[nt], false, false);
    }
    __syncthreads();
  }

  // epilogue
  #pragma unroll
  for (int nt = 0; nt < 4; ++nt) {
    int n = n0 + wn * 64 + nt * 16 + (lane & 15);
    if (n >= N) continue;
    float bv = bias ? bias[n] : 0.0f;
    #pragma unroll
    for (int r = 0; r < 8; ++r) {
      int m = m0 + wm * 16 + r + 8 * (lane >> 4);
      if (m >= M) continue;
      float v = acc[nt][r] + bv;
      if (Cf) Cf[(size_t)m * N + n] = v;
      else    Cb[(size_t)m * N + n] = (bf16)v;
    }
  }
}

// ---------------------------------------------------------------------------
// Precompute DETR sine position encoding (NT_ x F_)
// ---------------------------------------------------------------------------
__global__ __launch_bounds__(64) void pos_kernel(float* __restrict__ pos)
{
  int t = threadIdx.x;                 // 0..63
  int i = t >> 3, j = t & 7;
  const float sc = 6.283185307179586f / (8.0f + 1e-6f);
  float yv = (float)(i + 1) * sc;
  float xv = (float)(j + 1) * sc;
  for (int c = 0; c < F_; ++c) {
    int grp = c >> 6;                  // 0: y half, 1: x half
    int k   = c & 63;
    float base = grp ? xv : yv;
    float dt   = __powf(10000.0f, (float)(2 * (k >> 1)) * (1.0f / 64.0f));
    float p    = base / dt;
    pos[t * F_ + c] = (k & 1) ? __cosf(p) : __sinf(p);
  }
}

__device__ inline int regcls(int v) { return (v < 120) ? 0 : (v < 124 ? 1 : 2); }

__global__ __launch_bounds__(64) void mask_kernel(float* __restrict__ mask)
{
  int wdx = blockIdx.x;                // 0..255
  int t1  = threadIdx.x;               // 0..63
  int wh = wdx >> 4, ww = wdx & 15;
  int c1 = regcls(wh * 8 + (t1 >> 3)) * 3 + regcls(ww * 8 + (t1 & 7));
  for (int t2 = 0; t2 < NT_; ++t2) {
    int c2 = regcls(wh * 8 + (t2 >> 3)) * 3 + regcls(ww * 8 + (t2 & 7));
    mask[((size_t)wdx * NT_ + t1) * NT_ + t2] = (c1 != c2) ? -100.0f : 0.0f;
  }
}

// ---------------------------------------------------------------------------
// Weight converts: f32 -> bf16 (plain and transposed (O,I)->(I,O))
// ---------------------------------------------------------------------------
__global__ void cvt_bf16_kernel(const float* __restrict__ s, bf16* __restrict__ d, int n)
{
  int i = blockIdx.x * 256 + threadIdx.x;
  if (i < n) d[i] = (bf16)s[i];
}
__global__ void cvtT_bf16_kernel(const float* __restrict__ s, bf16* __restrict__ d, int O, int I)
{
  int idx = blockIdx.x * 256 + threadIdx.x;
  if (idx >= O * I) return;
  int o = idx / I, i = idx % I;
  d[(size_t)i * O + o] = (bf16)s[idx];
}

// ---------------------------------------------------------------------------
// Shift + window partition + LN1 + pos-add -> bf16 rows for the QKV GEMM.
// One wave per token; 4 channels per lane.
// ---------------------------------------------------------------------------
__global__ __launch_bounds__(128) void ln1_pos_kernel(
    const float* __restrict__ x, const float* __restrict__ g,
    const float* __restrict__ bb, const float* __restrict__ pos,
    bf16* __restrict__ Aout)
{
  int gtok = blockIdx.x * 4 + (threadIdx.x >> 5);
  int lane = threadIdx.x & 31;
  int t   = gtok & 63;
  int wdx = (gtok >> 6) & 255;
  int be  = gtok >> 14;                 // b*3+e, 0..5
  int wh = wdx >> 4, ww = wdx & 15;
  int i  = t >> 3,  j  = t & 7;
  int hh = (wh * 8 + i + SS_) & (H_ - 1);
  int wc = (ww * 8 + j + SS_) & (W_ - 1);
  const float* xp = x + (size_t)be * F_ * HW_ + hh * W_ + wc;

  float v[4]; float s = 0.0f;
  #pragma unroll
  for (int q = 0; q < 4; ++q) { int c = lane * 4 + q; v[q] = xp[(size_t)c * HW_]; s += v[q]; }
  for (int o = 16; o > 0; o >>= 1) s += __shfl_xor(s, o, 32);
  float mean = s * (1.0f / F_);
  float vs = 0.0f;
  #pragma unroll
  for (int q = 0; q < 4; ++q) { float d = v[q] - mean; vs += d * d; }
  for (int o = 16; o > 0; o >>= 1) vs += __shfl_xor(vs, o, 32);
  float rstd = rsqrtf(vs * (1.0f / F_) + 1e-5f);
  #pragma unroll
  for (int q = 0; q < 4; ++q) {
    int c = lane * 4 + q;
    float ov = (v[q] - mean) * rstd * g[c] + bb[c] + pos[t * F_ + c];
    Aout[(size_t)gtok * F_ + c] = (bf16)ov;
  }
}

// ---------------------------------------------------------------------------
// Cross-element shifted-window attention. One wave per (pair, window, head).
// qkv rows: [(b*3+e)*NW + wdx][t][3F], channel = p*F + d*NH + h (head inner).
// Output channel in concat: pair*F + h*16 + d (head outer).
// Qs/Ks row-major [t][d]; V stored transposed Vt[d][t]; P padded row-major.
// P/Vt fragments load as pairs of ds_load_b128; Q/K as one b128 plus zeros.
// ---------------------------------------------------------------------------
__global__ __launch_bounds__(32) void attn_kernel(
    const bf16* __restrict__ qkv, const float* __restrict__ mask,
    bf16* __restrict__ acat)
{
  const int h    = blockIdx.x;   // head 0..7
  const int bn   = blockIdx.y;   // 0..511 : b*NW + wdx
  const int pair = blockIdx.z;   // 0..5
  const int b    = bn >> 8, wdx = bn & 255;
  const int lane = threadIdx.x;

  __shared__ bf16  Qs[NT_ * DH_];
  __shared__ bf16  Ks[NT_ * DH_];
  __shared__ bf16  Vt[DH_ * VTP];
  __shared__ float Ss[NT_ * NT_];
  __shared__ bf16  Ps[NT_ * PSP];

  size_t qrow = (size_t)(b * 3 + d_QE[pair]) * NW_ + wdx;
  size_t krow = (size_t)(b * 3 + d_KE[pair]) * NW_ + wdx;
  for (int i = lane; i < NT_ * DH_; i += 32) {
    int t = i >> 4, d = i & 15;
    int ch = d * NH_ + h;
    Qs[i] = qkv[(qrow * NT_ + t) * C3_ + ch];
    Ks[i] = qkv[(krow * NT_ + t) * C3_ + F_ + ch];
    Vt[d * VTP + t] = qkv[(krow * NT_ + t) * C3_ + 2 * F_ + ch];
  }
  __syncthreads();

  const float scale = 0.25f;  // dh^-0.5
  const float* mw = mask + (size_t)wdx * NT_ * NT_;

  // S = (Q*scale) K^T + mask  (K=16 zero-padded to 32)
  for (int mt = 0; mt < 4; ++mt) {
    v16bf af;
    {
      int m = mt * 16 + (lane & 15);
      int kh = lane >> 4;
      frag_cvt c;
      v4u z = {};
      c.u[0] = *reinterpret_cast<const v4u*>(Qs + m * DH_ + kh * 8);
      c.u[1] = z;                       // k >= 16 padding
      af = c.v;
    }
    for (int nt = 0; nt < 4; ++nt) {
      v16bf bq = {};
      {                                 // elements 8..15 carry k>=16 (zero pad)
        int n = nt * 16 + (lane & 15);
        frag_cvt c;
        v4u z = {};
        c.u[0] = *reinterpret_cast<const v4u*>(Ks + n * DH_ + (lane >> 4) * 8);
        c.u[1] = z;
        bq = c.v;
      }
      v8f acc = {};
      acc = __builtin_amdgcn_wmma_f32_16x16x32_bf16(false, af, false, bq,
                                                    (short)0, acc, false, false);
      int nn = nt * 16 + (lane & 15);
      #pragma unroll
      for (int r = 0; r < 8; ++r) {
        int mm = mt * 16 + r + 8 * (lane >> 4);
        Ss[mm * NT_ + nn] = acc[r] * scale + mw[mm * NT_ + nn];
      }
    }
  }
  __syncthreads();

  // softmax: 2 rows per lane
  for (int rr = 0; rr < 2; ++rr) {
    int row = lane + rr * 32;
    float mx = -1e30f;
    for (int c = 0; c < NT_; ++c) mx = fmaxf(mx, Ss[row * NT_ + c]);
    float sum = 0.0f;
    for (int c = 0; c < NT_; ++c) { float e = __expf(Ss[row * NT_ + c] - mx); sum += e; Ss[row * NT_ + c] = e; }
    float inv = 1.0f / sum;
    for (int c = 0; c < NT_; ++c) Ps[row * PSP + c] = (bf16)(Ss[row * NT_ + c] * inv);
  }
  __syncthreads();

  // O = P (64x64) @ V (64x16); two K-steps
  for (int mt = 0; mt < 4; ++mt) {
    v8f acc = {};
    for (int k0 = 0; k0 < NT_; k0 += 32) {
      v16bf af;
      {
        int m = mt * 16 + (lane & 15);
        int kh = lane >> 4;
        const bf16* p = Ps + m * PSP + k0 + kh * 8;
        frag_cvt c;
        c.u[0] = *reinterpret_cast<const v4u*>(p);
        c.u[1] = *reinterpret_cast<const v4u*>(p + 16);
        af = c.v;
      }
      v16bf bq;
      {
        int n = lane & 15;
        int kh = (lane >> 4) * 8;
        const bf16* p = Vt + n * VTP + k0 + kh;
        frag_cvt c;
        c.u[0] = *reinterpret_cast<const v4u*>(p);
        c.u[1] = *reinterpret_cast<const v4u*>(p + 16);
        bq = c.v;
      }
      acc = __builtin_amdgcn_wmma_f32_16x16x32_bf16(false, af, false, bq,
                                                    (short)0, acc, false, false);
    }
    int d = lane & 15;
    #pragma unroll
    for (int r = 0; r < 8; ++r) {
      int t = mt * 16 + r + 8 * (lane >> 4);
      acat[((size_t)bn * NT_ + t) * C6_ + pair * F_ + h * DH_ + d] = (bf16)acc[r];
    }
  }
}

// GEGLU gate: g = gelu(h11) * h12 (bf16 in/out)
__global__ void geglu_gate_kernel(const bf16* __restrict__ h11,
                                  const bf16* __restrict__ h12,
                                  bf16* __restrict__ g, int n)
{
  int i = blockIdx.x * 256 + threadIdx.x;
  if (i >= n) return;
  float a = (float)h11[i], b = (float)h12[i];
  g[i] = (bf16)(gelu_exact(a) * b);
}

// Window-reverse of fc2 output into channels-last image (bf16) for conv11 GEMM
__global__ void winrev_kernel(const float* __restrict__ m, bf16* __restrict__ mimg)
{
  int idx = blockIdx.x * 256 + threadIdx.x;      // over PIX_*C3_
  if (idx >= PIX_ * C3_) return;
  int ch = idx % C3_;
  int pix = idx / C3_;
  int wc = pix & 127, hh = (pix >> 7) & 127, b = pix >> 14;
  size_t row = ((size_t)b * NW_ + (hh >> 3) * 16 + (wc >> 3)) * NT_ + (hh & 7) * 8 + (wc & 7);
  mimg[idx] = (bf16)m[row * C3_ + ch];
}

// reverse-shift + residual, LayerNorm2, channels-last bf16 copy of x for LCE.
// One wave per pixel; 12 channels per lane.
__global__ __launch_bounds__(128) void resid_ln2_kernel(
    const float* __restrict__ x, const float* __restrict__ convout,
    const float* __restrict__ g, const float* __restrict__ bb,
    float* __restrict__ xr, bf16* __restrict__ xcl, bf16* __restrict__ ln2o)
{
  int pix  = blockIdx.x * 4 + (threadIdx.x >> 5);
  int lane = threadIdx.x & 31;
  int wc = pix & 127, hh = (pix >> 7) & 127, b = pix >> 14;
  int hs = (hh - SS_) & (H_ - 1), ws2 = (wc - SS_) & (W_ - 1);
  const float* cop = convout + (((size_t)b * H_ + hs) * W_ + ws2) * C3_;

  float v[12]; float s = 0.0f;
  #pragma unroll
  for (int q = 0; q < 12; ++q) {
    int ch = lane * 12 + q;
    float xv = x[((size_t)b * 3 + (ch >> 7)) * F_ * HW_ + (size_t)(ch & 127) * HW_ + hh * W_ + wc];
    float r = xv + cop[ch];
    v[q] = r; s += r;
    xr[(size_t)pix * C3_ + ch] = r;
    xcl[(size_t)pix * C3_ + ch] = (bf16)xv;
  }
  for (int o = 16; o > 0; o >>= 1) s += __shfl_xor(s, o, 32);
  float mean = s * (1.0f / C3_);
  float vs = 0.0f;
  #pragma unroll
  for (int q = 0; q < 12; ++q) { float d = v[q] - mean; vs += d * d; }
  for (int o = 16; o > 0; o >>= 1) vs += __shfl_xor(vs, o, 32);
  float rstd = rsqrtf(vs * (1.0f / C3_) + 1e-5f);
  #pragma unroll
  for (int q = 0; q < 12; ++q) {
    int ch = lane * 12 + q;
    ln2o[(size_t)pix * C3_ + ch] = (bf16)((v[q] - mean) * rstd * g[ch] + bb[ch]);
  }
}

// depthwise 3x3 (SAME, zero pad) on 1536-ch hidden + GEGLU gate -> bf16 (768 ch)
__global__ void dwgate_kernel(const float* __restrict__ hid,
                              const float* __restrict__ dww,
                              bf16* __restrict__ gg)
{
  int idx = blockIdx.x * 256 + threadIdx.x;      // over PIX_*C6_
  if (idx >= PIX_ * C6_) return;
  int c = idx % C6_;
  int pix = idx / C6_;
  int wc = pix & 127, hh = (pix >> 7) & 127, b = pix >> 14;
  float a1 = 0.0f, a2 = 0.0f;
  for (int dy = -1; dy <= 1; ++dy)
    for (int dx = -1; dx <= 1; ++dx) {
      int y = hh + dy, xq = wc + dx;
      if (y < 0 || y > 127 || xq < 0 || xq > 127) continue;
      size_t po = (((size_t)b * H_ + y) * W_ + xq) * C12_;
      int wi = (dy + 1) * 3 + (dx + 1);
      a1 += hid[po + c]        * dww[c * 9 + wi];
      a2 += hid[po + C6_ + c]  * dww[(c + C6_) * 9 + wi];
    }
  gg[(size_t)pix * C6_ + c] = (bf16)(gelu_exact(a1) * a2);
}

// LCE 3x3 dense conv 48->48 + bias
__global__ void lce_conv3_kernel(const float* __restrict__ l1,
                                 const float* __restrict__ w2,
                                 const float* __restrict__ b2,
                                 float* __restrict__ l2o)
{
  int idx = blockIdx.x * 256 + threadIdx.x;      // over PIX_*48
  if (idx >= PIX_ * 48) return;
  int oc = idx % 48;
  int pix = idx / 48;
  int wc = pix & 127, hh = (pix >> 7) & 127, b = pix >> 14;
  float a = b2[oc];
  for (int dy = -1; dy <= 1; ++dy)
    for (int dx = -1; dx <= 1; ++dx) {
      int y = hh + dy, xq = wc + dx;
      if (y < 0 || y > 127 || xq < 0 || xq > 127) continue;
      size_t po = (((size_t)b * H_ + y) * W_ + xq) * 48;
      const float* wp = w2 + ((size_t)oc * 48) * 9 + (dy + 1) * 3 + (dx + 1);
      for (int ic = 0; ic < 48; ++ic) a += l1[po + ic] * wp[(size_t)ic * 9];
    }
  l2o[(size_t)pix * 48 + oc] = a;
}

// LCE 1x1 conv 48->384 + bias + LeakyReLU(0.2)
__global__ void lce_l3_kernel(const float* __restrict__ l2v,
                              const float* __restrict__ w3,
                              const float* __restrict__ b3,
                              float* __restrict__ lout)
{
  int idx = blockIdx.x * 256 + threadIdx.x;      // over PIX_*C3_
  if (idx >= PIX_ * C3_) return;
  int oc = idx % C3_;
  int pix = idx / C3_;
  const float* lp = l2v + (size_t)pix * 48;
  float a = b3[oc];
  for (int ic = 0; ic < 48; ++ic) a += lp[ic] * w3[(size_t)oc * 48 + ic];
  lout[idx] = (a >= 0.0f) ? a : 0.2f * a;
}

__global__ __launch_bounds__(256) void lce_mean_kernel(const float* __restrict__ lout,
                                                       float* __restrict__ lmean)
{
  int ch = blockIdx.x;   // 0..383
  int b  = blockIdx.y;   // 0..1
  float s = 0.0f;
  for (int p = threadIdx.x; p < HW_; p += 256)
    s += lout[((size_t)b * HW_ + p) * C3_ + ch];
  __shared__ float red[256];
  red[threadIdx.x] = s; __syncthreads();
  for (int o = 128; o > 0; o >>= 1) {
    if (threadIdx.x < o) red[threadIdx.x] += red[threadIdx.x + o];
    __syncthreads();
  }
  if (threadIdx.x == 0) lmean[b * C3_ + ch] = red[0] * (1.0f / HW_);
}

__global__ __launch_bounds__(64) void lce_se_kernel(const float* __restrict__ lmean,
                                                    const float* __restrict__ fc1,
                                                    const float* __restrict__ fc2,
                                                    float* __restrict__ svec)
{
  int b = blockIdx.x;
  __shared__ float t1[48];
  const float* mv = lmean + b * C3_;
  for (int j = threadIdx.x; j < 48; j += 64) {
    float a = 0.0f;
    for (int c = 0; c < C3_; ++c) a += mv[c] * fc1[c * 48 + j];
    t1[j] = fmaxf(a, 0.0f);
  }
  __syncthreads();
  for (int ch = threadIdx.x; ch < C3_; ch += 64) {
    float a = 0.0f;
    for (int j = 0; j < 48; ++j) a += t1[j] * fc2[j * C3_ + ch];
    svec[b * C3_ + ch] = 1.0f / (1.0f + __expf(-a));
  }
}

__global__ void final_kernel(const float* __restrict__ xr, const float* __restrict__ y,
                             const float* __restrict__ lout, const float* __restrict__ svec,
                             float* __restrict__ out)
{
  int idx = blockIdx.x * 256 + threadIdx.x;      // over B*E*F*H*W
  if (idx >= B_ * E_ * F_ * HW_) return;
  int wc = idx & 127, hh = (idx >> 7) & 127;
  int c  = (idx >> 14) & 127;
  int be = idx >> 21;                             // b*3+e
  int b  = be / 3;
  int ch = (be % 3) * F_ + c;
  size_t pix = ((size_t)b * H_ + hh) * W_ + wc;
  out[idx] = xr[pix * C3_ + ch] + y[pix * C3_ + ch] + lout[pix * C3_ + ch] * svec[b * C3_ + ch];
}

// ---------------------------------------------------------------------------
static void launch_gemm(const bf16* A, const bf16* Bm, const float* bias,
                        float* Cf, bf16* Cb, int M, int N, int K, hipStream_t s)
{
  dim3 grid((unsigned)((N + 127) / 128), (unsigned)((M + 63) / 64));
  gemm_bf16_kernel<<<grid, 256, 0, s>>>(A, Bm, bias, Cf, Cb, M, N, K);
}

extern "C" void kernel_launch(void* const* d_in, const int* in_sizes, int n_in,
                              void* d_out, int out_size, void* d_ws, size_t ws_size,
                              hipStream_t stream)
{
  (void)in_sizes; (void)n_in; (void)out_size; (void)ws_size;
  const float* x        = (const float*)d_in[0];
  const float* ln1_g    = (const float*)d_in[1];
  const float* ln1_b    = (const float*)d_in[2];
  const float* ln2_g    = (const float*)d_in[3];
  const float* ln2_b    = (const float*)d_in[4];
  const float* w_qkv    = (const float*)d_in[5];
  const float* fc11_w   = (const float*)d_in[6];
  const float* fc11_b   = (const float*)d_in[7];
  const float* fc12_w   = (const float*)d_in[8];
  const float* fc12_b   = (const float*)d_in[9];
  const float* fc2_w    = (const float*)d_in[10];
  const float* fc2_b    = (const float*)d_in[11];
  const float* conv11_w = (const float*)d_in[12];
  const float* ff_in_w  = (const float*)d_in[13];
  const float* ff_dw_w  = (const float*)d_in[14];
  const float* ff_out_w = (const float*)d_in[15];
  const float* lce_w1   = (const float*)d_in[16];
  const float* lce_b1   = (const float*)d_in[17];
  const float* lce_w2   = (const float*)d_in[18];
  const float* lce_b2   = (const float*)d_in[19];
  const float* lce_w3   = (const float*)d_in[20];
  const float* lce_b3   = (const float*)d_in[21];
  const float* lce_fc1  = (const float*)d_in[22];
  const float* lce_fc2  = (const float*)d_in[23];
  float* out = (float*)d_out;

  char* ws = (char*)d_ws;
  size_t off = 0;
  auto alloc = [&](size_t bytes) -> char* {
    char* p = ws + off;
    off += (bytes + 255) & ~(size_t)255;
    return p;
  };

  float* pos     = (float*)alloc((size_t)NT_ * F_ * 4);
  float* mask    = (float*)alloc((size_t)NW_ * NT_ * NT_ * 4);
  bf16* wqkv_bf  = (bf16*)alloc((size_t)F_ * C3_ * 2);
  bf16* fc11_bf  = (bf16*)alloc((size_t)C6_ * C12_ * 2);
  bf16* fc12_bf  = (bf16*)alloc((size_t)C6_ * C12_ * 2);
  bf16* fc2_bf   = (bf16*)alloc((size_t)C12_ * C3_ * 2);
  bf16* c11T_bf  = (bf16*)alloc((size_t)C3_ * C3_ * 2);
  bf16* ffinT_bf = (bf16*)alloc((size_t)C3_ * C12_ * 2);
  bf16* ffoutT_bf= (bf16*)alloc((size_t)C6_ * C3_ * 2);
  bf16* lw1T_bf  = (bf16*)alloc((size_t)C3_ * 48 * 2);
  bf16* A_ln     = (bf16*)alloc((size_t)TOK_ * F_ * 2);
  bf16* qkvb     = (bf16*)alloc((size_t)TOK_ * C3_ * 2);
  char* acat_m   = alloc((size_t)PIX_ * C6_ * 2);        // acat bf16, reused as m f32
  bf16* acat     = (bf16*)acat_m;
  float* mbuf    = (float*)acat_m;                       // PIX_*C3_*4 == same bytes
  bf16* h11      = (bf16*)alloc((size_t)PIX_ * C12_ * 2);
  bf16* h12      = (bf16*)alloc((size_t)PIX_ * C12_ * 2);
  float* hid     = (float*)h11;                          // spans h11+h12 (contiguous)
  bf16* gbuf     = (bf16*)alloc((size_t)PIX_ * C12_ * 2);
  bf16* mimg     = (bf16*)alloc((size_t)PIX_ * C3_ * 2);
  char* cvy      = alloc((size_t)PIX_ * C3_ * 4);        // convout f32, reused as y f32
  float* convout = (float*)cvy;
  float* ybuf    = (float*)cvy;
  float* xrbuf   = (float*)alloc((size_t)PIX_ * C3_ * 4);
  bf16* xcl      = (bf16*)alloc((size_t)PIX_ * C3_ * 2);
  bf16* ln2b     = (bf16*)alloc((size_t)PIX_ * C3_ * 2);
  bf16* ggbuf    = (bf16*)alloc((size_t)PIX_ * C6_ * 2);
  float* l1buf   = (float*)alloc((size_t)PIX_ * 48 * 4);
  float* l2buf   = (float*)alloc((size_t)PIX_ * 48 * 4);
  float* lbuf    = (float*)alloc((size_t)PIX_ * C3_ * 4);
  float* lmean   = (float*)alloc((size_t)B_ * C3_ * 4);
  float* svec    = (float*)alloc((size_t)B_ * C3_ * 4);

  // constants
  pos_kernel<<<1, 64, 0, stream>>>(pos);
  mask_kernel<<<NW_, 64, 0, stream>>>(mask);

  // weight conversion
  auto cvt  = [&](const float* s, bf16* d, int n) {
    cvt_bf16_kernel<<<(n + 255) / 256, 256, 0, stream>>>(s, d, n);
  };
  auto cvtT = [&](const float* s, bf16* d, int O, int I) {
    cvtT_bf16_kernel<<<(O * I + 255) / 256, 256, 0, stream>>>(s, d, O, I);
  };
  cvt(w_qkv,  wqkv_bf, F_ * C3_);
  cvt(fc11_w, fc11_bf, C6_ * C12_);
  cvt(fc12_w, fc12_bf, C6_ * C12_);
  cvt(fc2_w,  fc2_bf,  C12_ * C3_);
  cvtT(conv11_w, c11T_bf,  C3_,  C3_);
  cvtT(ff_in_w,  ffinT_bf, C12_, C3_);
  cvtT(ff_out_w, ffoutT_bf, C3_, C6_);
  cvtT(lce_w1,   lw1T_bf,  48,   C3_);

  // 1) shift + window + LN1 + pos  ->  A_ln
  ln1_pos_kernel<<<TOK_ / 4, 128, 0, stream>>>(x, ln1_g, ln1_b, pos, A_ln);
  // 2) QKV projection (bf16 out)
  launch_gemm(A_ln, wqkv_bf, nullptr, nullptr, qkvb, TOK_, C3_, F_, stream);
  // 3) attention (6 cross pairs) -> acat
  attn_kernel<<<dim3(NH_, B_ * NW_, 6), 32, 0, stream>>>(qkvb, mask, acat);
  // 4) GEGLU: fc11, fc12 (bf16 out), gate, fc2 (f32 out -> mbuf over acat)
  launch_gemm(acat, fc11_bf, fc11_b, nullptr, h11, PIX_, C12_, C6_, stream);
  launch_gemm(acat, fc12_bf, fc12_b, nullptr, h12, PIX_, C12_, C6_, stream);
  geglu_gate_kernel<<<(PIX_ * C12_ + 255) / 256, 256, 0, stream>>>(h11, h12, gbuf, PIX_ * C12_);
  launch_gemm(gbuf, fc2_bf, fc2_b, mbuf, nullptr, PIX_, C3_, C12_, stream);
  // 5) window reverse + 1x1 conv + reverse-shift + residual + LN2
  winrev_kernel<<<(PIX_ * C3_ + 255) / 256, 256, 0, stream>>>(mbuf, mimg);
  launch_gemm(mimg, c11T_bf, nullptr, convout, nullptr, PIX_, C3_, C3_, stream);
  resid_ln2_kernel<<<PIX_ / 4, 128, 0, stream>>>(x, convout, ln2_g, ln2_b, xrbuf, xcl, ln2b);
  // 6) conv-GEGLU FFN
  launch_gemm(ln2b, ffinT_bf, nullptr, hid, nullptr, PIX_, C12_, C3_, stream);
  dwgate_kernel<<<(PIX_ * C6_ + 255) / 256, 256, 0, stream>>>(hid, ff_dw_w, ggbuf);
  launch_gemm(ggbuf, ffoutT_bf, nullptr, ybuf, nullptr, PIX_, C3_, C6_, stream);
  // 7) LocalContextExtractor
  launch_gemm(xcl, lw1T_bf, lce_b1, l1buf, nullptr, PIX_, 48, C3_, stream);
  lce_conv3_kernel<<<(PIX_ * 48 + 255) / 256, 256, 0, stream>>>(l1buf, lce_w2, lce_b2, l2buf);
  lce_l3_kernel<<<(PIX_ * C3_ + 255) / 256, 256, 0, stream>>>(l2buf, lce_w3, lce_b3, lbuf);
  lce_mean_kernel<<<dim3(C3_, B_), 256, 0, stream>>>(lbuf, lmean);
  lce_se_kernel<<<B_, 64, 0, stream>>>(lmean, lce_fc1, lce_fc2, svec);
  // 8) final sum -> (b, e, f, h, w)
  final_kernel<<<(B_ * E_ * F_ * HW_ + 255) / 256, 256, 0, stream>>>(xrbuf, ybuf, lbuf, svec, out);
}
